// DRBM_84061099917740
// MI455X (gfx1250) — matrix operand order repaired and checked by reading, not compiled, not run on hardware
//
#include <hip/hip_runtime.h>
#include <hip/hip_bf16.h>
#include <stdint.h>

// Problem sizes (fixed by the reference)
#define B_SZ   2048
#define NVIS   2048
#define NHID   1024
#define NCLASS 64

typedef __attribute__((ext_vector_type(16))) __bf16 v16bf;
typedef __attribute__((ext_vector_type(8)))  __bf16 v8bf;
typedef __attribute__((ext_vector_type(8)))  float  v8f;
typedef __attribute__((ext_vector_type(4)))  unsigned int v4u;
typedef __attribute__((ext_vector_type(8)))  int v8i;
typedef __attribute__((ext_vector_type(4)))  int v4i;

#if __has_builtin(__builtin_amdgcn_tensor_load_to_lds)
#define HAVE_TDM 1
#else
#define HAVE_TDM 0
#endif
#if __has_include(<hip/amd_detail/amd_gfx1250_TDM.h>)
#define TDM_SIX_ARG 1   // therock-10.0 headers -> 6-arg builtin
#endif

// ---------------------------------------------------------------------------
// GEMM tiling: workgroup computes 64x128 of pre; 8 waves, each owns a 32x32
// sub-block = 2x2 WMMA tiles. K-step = 32 (bf16). Double-buffered LDS tiles
// are filled by the Tensor Data Mover (4 tensor_load_to_lds per step) and
// consumed via ds_load_b128 in the documented 16-bit A/B fragment layouts.
// ---------------------------------------------------------------------------
#define BM   64
#define BN   128
#define TK   32
#define LDA  40            // LDS row stride in bf16 (80B = 64B tile row + 16B pad)
#define NSTEPS (NVIS / TK) // 64

typedef uint32_t u32;

#if HAVE_TDM
// Issue one TDM 2-D tile load: tile_d1 rows x tile_d0 elems (2B each) from a
// row-major bf16 tensor with row stride `stride_elems`, into LDS at lds_addr
// with 16B of padding appended after every 64B row (-> 80B LDS row stride).
__device__ __forceinline__ void tdm_load_2d(u32 lds_addr, const void* gaddr,
                                            u32 tensor_d0, u32 tensor_d1,
                                            u32 tile_d0, u32 tile_d1,
                                            u32 stride_elems)
{
    uint64_t ga = (uint64_t)(uintptr_t)gaddr;
    v4u g0;
    g0[0] = 1u;                                    // count=1, user mode
    g0[1] = lds_addr;                              // LDS byte address
    g0[2] = (u32)ga;                               // global_addr[31:0]
    g0[3] = (u32)((ga >> 32) & 0x01FFFFFFu)        // global_addr[56:32]
          | (2u << 30);                            // type = 2 ("image")
    v8i g1;
    g1[0] = (int)((1u << 16)                       // data_size = 2 bytes
                | (1u << 20)                       // pad_enable
                | (3u << 22)                       // pad_interval: 16 DWORDs (64B)
                | (3u << 25));                     // pad_amount: 4 DWORDs (16B)
    g1[1] = (int)((tensor_d0 & 0xFFFFu) << 16);    // atomic_barrier=0 | dim0 lo
    g1[2] = (int)(((tensor_d0 >> 16) & 0xFFFFu) | ((tensor_d1 & 0xFFFFu) << 16));
    g1[3] = (int)(((tensor_d1 >> 16) & 0xFFFFu) | (tile_d0 << 16));
    g1[4] = (int)(tile_d1 & 0xFFFFu);              // tile_dim1 | tile_dim2=0
    g1[5] = (int)stride_elems;                     // tensor_dim0_stride[31:0]
    g1[6] = 0;                                     // stride hi / dim1_stride
    g1[7] = 0;
    v4i z4 = {0, 0, 0, 0};
#ifdef TDM_SIX_ARG
    v8i z8 = {0, 0, 0, 0, 0, 0, 0, 0};
    __builtin_amdgcn_tensor_load_to_lds(g0, g1, z4, z4, z8, 0);
#else
    __builtin_amdgcn_tensor_load_to_lds(g0, g1, z4, z4, 0);
#endif
}
#endif

// ---------------------------------------------------------------------------
// Pre-pass A: v (f32) -> bf16 hi/lo, same layout.
// ---------------------------------------------------------------------------
__global__ __launch_bounds__(256) void convert_v_kernel(
    const float* __restrict__ v, __bf16* __restrict__ vhi, __bf16* __restrict__ vlo)
{
    size_t base = (size_t)blockIdx.x * 1024 + threadIdx.x;
    #pragma unroll
    for (int u = 0; u < 4; ++u) {
        size_t i = base + (size_t)u * 256;
        float x = v[i];
        __bf16 h = (__bf16)x;
        vhi[i] = h;
        vlo[i] = (__bf16)(x - (float)h);
    }
}

// ---------------------------------------------------------------------------
// Pre-pass B: W [NVIS][NHID] f32 -> Wt [NHID][NVIS] bf16 hi/lo (transposed so
// the GEMM's B tiles are K-contiguous and TDM/ds_load_b128-friendly).
// ---------------------------------------------------------------------------
__global__ __launch_bounds__(256) void convert_Wt_kernel(
    const float* __restrict__ W, __bf16* __restrict__ Wthi, __bf16* __restrict__ Wtlo)
{
    __shared__ float tile[32][33];
    const int k0 = blockIdx.x * 32;          // along NVIS
    const int n0 = blockIdx.y * 32;          // along NHID
    const int tx = threadIdx.x & 31, ty = threadIdx.x >> 5;
    #pragma unroll
    for (int r = 0; r < 4; ++r) {
        int k = ty + r * 8;
        tile[k][tx] = W[(size_t)(k0 + k) * NHID + n0 + tx];
    }
    __syncthreads();
    #pragma unroll
    for (int r = 0; r < 4; ++r) {
        int n = ty + r * 8;
        float x = tile[tx][n];               // = W[k0+tx][n0+n]
        __bf16 h = (__bf16)x;
        size_t o = (size_t)(n0 + n) * NVIS + k0 + tx;
        Wthi[o] = h;
        Wtlo[o] = (__bf16)(x - (float)h);
    }
}

// ---------------------------------------------------------------------------
// Kernel 1: pre = v@W + c via bf16x3-split WMMA, TDM-fed double-buffered LDS.
// ---------------------------------------------------------------------------
__global__ __launch_bounds__(256) void drbm_gemm_kernel(
    const __bf16* __restrict__ vhi, const __bf16* __restrict__ vlo,
    const __bf16* __restrict__ Wthi, const __bf16* __restrict__ Wtlo,
    const float* __restrict__ c, float* __restrict__ pre)
{
    __shared__ __align__(16) __bf16 As_hi[2][BM][LDA];
    __shared__ __align__(16) __bf16 As_lo[2][BM][LDA];
    __shared__ __align__(16) __bf16 Bs_hi[2][BN][LDA];   // [n][k] (Wt rows)
    __shared__ __align__(16) __bf16 Bs_lo[2][BN][LDA];

    const int tid  = threadIdx.x;
    const int wave = tid >> 5;
    const int lane = tid & 31;
    const int half = lane >> 4;
    const int l16  = lane & 15;
    const int row0 = blockIdx.y * BM;
    const int col0 = blockIdx.x * BN;
    const int wrow = (wave >> 2) * 32;       // wave's rows within BM
    const int wcol = (wave & 3) * 32;        // wave's cols within BN

    const __bf16* gAh = vhi  + (size_t)row0 * NVIS;
    const __bf16* gAl = vlo  + (size_t)row0 * NVIS;
    const __bf16* gBh = Wthi + (size_t)col0 * NVIS;
    const __bf16* gBl = Wtlo + (size_t)col0 * NVIS;

    v8f acc[2][2] = {};

    // Fragment addressing per documented 16-bit WMMA layouts:
    // A (16x32): lane<16 -> K{0..7,16..23}; lane>=16 -> K{8..15,24..31}
    // B (32x16): lane<16 -> K{0..7,16..23}; lane>=16 -> K{8..15,24..31}; N = lane%16
    const int kA = half * 8;
    const int kB = half * 8;

#if HAVE_TDM
    if (wave == 0) {
        tdm_load_2d((u32)(uintptr_t)&As_hi[0][0][0], gAh, NVIS, B_SZ, TK, BM, NVIS);
        tdm_load_2d((u32)(uintptr_t)&As_lo[0][0][0], gAl, NVIS, B_SZ, TK, BM, NVIS);
        tdm_load_2d((u32)(uintptr_t)&Bs_hi[0][0][0], gBh, NVIS, NHID, TK, BN, NVIS);
        tdm_load_2d((u32)(uintptr_t)&Bs_lo[0][0][0], gBl, NVIS, NHID, TK, BN, NVIS);
    }
#else
    {   // fallback: vectorized cooperative staging of buffer 0
        const int ar = tid >> 2, ap = (tid & 3) * 8;            // 64 rows x 4 chunks
        *(v8bf*)&As_hi[0][ar][ap] = *(const v8bf*)(gAh + (size_t)ar * NVIS + ap);
        *(v8bf*)&As_lo[0][ar][ap] = *(const v8bf*)(gAl + (size_t)ar * NVIS + ap);
        #pragma unroll
        for (int q = 0; q < 2; ++q) {                           // 128 rows x 4 chunks
            int id = tid + q * 256, br = id >> 2, bp = (id & 3) * 8;
            *(v8bf*)&Bs_hi[0][br][bp] = *(const v8bf*)(gBh + (size_t)br * NVIS + bp);
            *(v8bf*)&Bs_lo[0][br][bp] = *(const v8bf*)(gBl + (size_t)br * NVIS + bp);
        }
        __syncthreads();
    }
#endif

    #pragma unroll 1
    for (int i = 0; i < NSTEPS; ++i) {
        const int buf = i & 1;
        const int nb  = buf ^ 1;
        const int kn  = (i + 1) * TK;

#if HAVE_TDM
        if (wave == 0) {
            if (i + 1 < NSTEPS) {
                tdm_load_2d((u32)(uintptr_t)&As_hi[nb][0][0], gAh + kn, NVIS, B_SZ, TK, BM, NVIS);
                tdm_load_2d((u32)(uintptr_t)&As_lo[nb][0][0], gAl + kn, NVIS, B_SZ, TK, BM, NVIS);
                tdm_load_2d((u32)(uintptr_t)&Bs_hi[nb][0][0], gBh + kn, NVIS, NHID, TK, BN, NVIS);
                tdm_load_2d((u32)(uintptr_t)&Bs_lo[nb][0][0], gBl + kn, NVIS, NHID, TK, BN, NVIS);
                __builtin_amdgcn_s_wait_tensorcnt(4);   // current buf's 4 done
            } else {
                __builtin_amdgcn_s_wait_tensorcnt(0);
            }
        }
        __syncthreads();                                 // publish buf to all waves
#else
        if (i + 1 < NSTEPS) {
            const int ar = tid >> 2, ap = (tid & 3) * 8;
            *(v8bf*)&As_hi[nb][ar][ap] = *(const v8bf*)(gAh + (size_t)ar * NVIS + kn + ap);
            *(v8bf*)&As_lo[nb][ar][ap] = *(const v8bf*)(gAl + (size_t)ar * NVIS + kn + ap);
            #pragma unroll
            for (int q = 0; q < 2; ++q) {
                int id = tid + q * 256, br = id >> 2, bp = (id & 3) * 8;
                *(v8bf*)&Bs_hi[nb][br][bp] = *(const v8bf*)(gBh + (size_t)br * NVIS + kn + bp);
                *(v8bf*)&Bs_lo[nb][br][bp] = *(const v8bf*)(gBl + (size_t)br * NVIS + kn + bp);
            }
        }
#endif

        union bfu { v8bf h[2]; v16bf w; };
        bfu ah[2], al[2], bh[2], bl[2];
        #pragma unroll
        for (int rt = 0; rt < 2; ++rt) {
            const int r = wrow + rt * 16 + l16;
            ah[rt].h[0] = *(const v8bf*)&As_hi[buf][r][kA];
            ah[rt].h[1] = *(const v8bf*)&As_hi[buf][r][kA + 16];
            al[rt].h[0] = *(const v8bf*)&As_lo[buf][r][kA];
            al[rt].h[1] = *(const v8bf*)&As_lo[buf][r][kA + 16];
        }
        #pragma unroll
        for (int ct = 0; ct < 2; ++ct) {
            const int n = wcol + ct * 16 + l16;
            bh[ct].h[0] = *(const v8bf*)&Bs_hi[buf][n][kB];
            bh[ct].h[1] = *(const v8bf*)&Bs_hi[buf][n][kB + 16];
            bl[ct].h[0] = *(const v8bf*)&Bs_lo[buf][n][kB];
            bl[ct].h[1] = *(const v8bf*)&Bs_lo[buf][n][kB + 16];
        }

        // bf16x3 split: hi*hi + hi*lo + lo*hi (~f32 accuracy), f32 accumulate
        #pragma unroll
        for (int rt = 0; rt < 2; ++rt)
            #pragma unroll
            for (int ct = 0; ct < 2; ++ct) {
                acc[rt][ct] = __builtin_amdgcn_wmma_f32_16x16x32_bf16(
                    false, ah[rt].w, false, bh[ct].w, (short)0, acc[rt][ct], false, false);
                acc[rt][ct] = __builtin_amdgcn_wmma_f32_16x16x32_bf16(
                    false, ah[rt].w, false, bl[ct].w, (short)0, acc[rt][ct], false, false);
                acc[rt][ct] = __builtin_amdgcn_wmma_f32_16x16x32_bf16(
                    false, al[rt].w, false, bh[ct].w, (short)0, acc[rt][ct], false, false);
            }
        __syncthreads();                                 // done reading buf
    }

    // C/D layout: lane L, VGPR r -> M = r + 8*(L/16), N = L%16
    #pragma unroll
    for (int ct = 0; ct < 2; ++ct) {
        const int col = col0 + wcol + ct * 16 + l16;
        const float bias = c[col];
        #pragma unroll
        for (int rt = 0; rt < 2; ++rt)
            #pragma unroll
            for (int r = 0; r < 8; ++r) {
                int row = row0 + wrow + rt * 16 + r + 8 * half;
                pre[(size_t)row * NHID + col] = acc[rt][ct][r] + bias;
            }
    }
}

// ---------------------------------------------------------------------------
// Kernel 2: F[b,y] = d[y] + sum_j softplus(pre[b,j] + U[y,j]); softmax over
// classes; argmax -> one-hot. 4 rows/WG amortizes U reads; wave w owns 8
// classes; lanes stride hidden dim; 2 TRANS ops per softplus.
// ---------------------------------------------------------------------------
#define K2_ROWS 4

__global__ __launch_bounds__(256) void drbm_energy_kernel(
    const float* __restrict__ pre, const float* __restrict__ U,
    const float* __restrict__ dvec, float* __restrict__ probs,
    int* __restrict__ onehot)
{
    __shared__ float sPre[K2_ROWS][NHID];
    __shared__ float sF[K2_ROWS][NCLASS];

    const int tid  = threadIdx.x;
    const int wave = tid >> 5;
    const int lane = tid & 31;
    const int b0   = blockIdx.x * K2_ROWS;
    const int ybase = wave * 8;

    #pragma unroll
    for (int r = 0; r < K2_ROWS; ++r)
        for (int j = tid; j < NHID; j += 256)
            sPre[r][j] = pre[(size_t)(b0 + r) * NHID + j];
    __syncthreads();

    float acc[K2_ROWS][8];
    #pragma unroll
    for (int r = 0; r < K2_ROWS; ++r)
        #pragma unroll
        for (int yy = 0; yy < 8; ++yy) acc[r][yy] = 0.0f;

    for (int jj = 0; jj < NHID; jj += 32) {
        const int j = jj + lane;
        float p[K2_ROWS];
        #pragma unroll
        for (int r = 0; r < K2_ROWS; ++r) p[r] = sPre[r][j];
        #pragma unroll
        for (int yy = 0; yy < 8; ++yy) {
            float u = U[(size_t)(ybase + yy) * NHID + j];   // L2/WGP$-resident
            #pragma unroll
            for (int r = 0; r < K2_ROWS; ++r) {
                float x = p[r] + u;
                float t = __expf(-fabsf(x));
                acc[r][yy] += fmaxf(x, 0.0f) + __logf(1.0f + t);
            }
        }
    }

    #pragma unroll
    for (int r = 0; r < K2_ROWS; ++r) {
        #pragma unroll
        for (int yy = 0; yy < 8; ++yy) {
            float s = acc[r][yy];
            #pragma unroll
            for (int off = 16; off > 0; off >>= 1)
                s += __shfl_xor(s, off, 32);
            if (lane == 0) sF[r][ybase + yy] = s + dvec[ybase + yy];
        }
    }
    __syncthreads();

    if (wave < K2_ROWS) {
        const int b = b0 + wave;
        float f0 = sF[wave][lane];
        float f1 = sF[wave][lane + 32];
        float m = fmaxf(f0, f1);
        #pragma unroll
        for (int off = 16; off > 0; off >>= 1)
            m = fmaxf(m, __shfl_xor(m, off, 32));
        float e0 = __expf(f0 - m);
        float e1 = __expf(f1 - m);
        float s = e0 + e1;
        #pragma unroll
        for (int off = 16; off > 0; off >>= 1)
            s += __shfl_xor(s, off, 32);
        float inv = 1.0f / s;
        probs[(size_t)b * NCLASS + lane]      = e0 * inv;
        probs[(size_t)b * NCLASS + lane + 32] = e1 * inv;

        float mv = (f0 >= f1) ? f0 : f1;
        int   mi = (f0 >= f1) ? lane : lane + 32;
        #pragma unroll
        for (int off = 16; off > 0; off >>= 1) {
            float ov = __shfl_xor(mv, off, 32);
            int   oi = __shfl_xor(mi, off, 32);
            if (ov > mv || (ov == mv && oi < mi)) { mv = ov; mi = oi; }
        }
        onehot[(size_t)b * NCLASS + lane]      = (mi == lane)      ? 1 : 0;
        onehot[(size_t)b * NCLASS + lane + 32] = (mi == lane + 32) ? 1 : 0;
    }
}

// ---------------------------------------------------------------------------
extern "C" void kernel_launch(void* const* d_in, const int* in_sizes, int n_in,
                              void* d_out, int out_size, void* d_ws, size_t ws_size,
                              hipStream_t stream) {
    const float* v = (const float*)d_in[0];   // [B, NVIS]
    const float* W = (const float*)d_in[1];   // [NVIS, NHID]
    const float* c = (const float*)d_in[2];   // [1, NHID]
    const float* d = (const float*)d_in[3];   // [1, NCLASS]
    const float* U = (const float*)d_in[4];   // [NCLASS, NHID]

    char* ws = (char*)d_ws;
    float*  pre  = (float*) (ws);                     // 8 MB  [B, NHID] f32
    __bf16* vhi  = (__bf16*)(ws + (8u  << 20));       // 8 MB  [B, NVIS] bf16
    __bf16* vlo  = (__bf16*)(ws + (16u << 20));       // 8 MB
    __bf16* Wthi = (__bf16*)(ws + (24u << 20));       // 4 MB  [NHID, NVIS] bf16
    __bf16* Wtlo = (__bf16*)(ws + (28u << 20));       // 4 MB  (32 MB total)

    float* probs  = (float*)d_out;                                  // [B, NCLASS] f32
    int*   onehot = (int*)((float*)d_out + (size_t)B_SZ * NCLASS);  // [B, NCLASS] i32

    convert_v_kernel<<<(B_SZ * NVIS) / 1024, 256, 0, stream>>>(v, vhi, vlo);
    convert_Wt_kernel<<<dim3(NVIS / 32, NHID / 32), 256, 0, stream>>>(W, Wthi, Wtlo);

    dim3 g1(NHID / BN, B_SZ / BM);   // (8, 32)
    drbm_gemm_kernel<<<g1, 256, 0, stream>>>(vhi, vlo, Wthi, Wtlo, c, pre);

    drbm_energy_kernel<<<B_SZ / K2_ROWS, 256, 0, stream>>>(pre, U, d, probs, onehot);
}
